// TrajRVQTransformer_88021059764285
// MI455X (gfx1250) — matrix operand order repaired and checked by reading, not compiled, not run on hardware
//
#include <hip/hip_runtime.h>
#include <hip/hip_bf16.h>

typedef __attribute__((ext_vector_type(16))) _Float16 v16h;
typedef __attribute__((ext_vector_type(8)))  _Float16 v8h;
typedef __attribute__((ext_vector_type(8)))  float    v8f;

#define NB   32768           // batch
#define DM   256             // d_model
#define NQV  1024            // vocab
#define NQS  10              // quantizers

// ---------------------------------------------------------------------------
// WMMA GEMM:  C[M,N] = A[M,K] (f16, row-major, lda) x W[N,K]^T (f16, row-major, ldb)
// One wave -> (16*TM) x (16*TN) output block; all A/B fragments for a K-step
// are loaded up-front (distinct registers) so the 12 b128 loads issue as one
// clause and WMMAs start on partial loadcnt waits. f32 accum, fused epilogues.
// EP: 0 outH=acc+b | 1 outH=relu(acc+b) | 2 outF+=acc+b | 3 outF=acc+b?+b2?
//     4 outF=acc+b and outH=f16(same)   | 5 outF=b - 2*acc   (VQ distance)
// ---------------------------------------------------------------------------
template<int EP, int TM, int TN>
__global__ __launch_bounds__(256)
void wmma_gemm(const _Float16* __restrict__ A, int lda,
               const _Float16* __restrict__ W, int ldb,
               const float* __restrict__ bias, const float* __restrict__ bias2,
               float* __restrict__ outF, _Float16* __restrict__ outH,
               int M, int N, int K)
{
    const int lane = threadIdx.x & 31;
    const int wid  = threadIdx.x >> 5;
    const int ntg  = N / (16 * TN);
    const int tile = blockIdx.x * 8 + wid;
    if (tile >= (M / (16 * TM)) * ntg) return;
    const int tm  = (tile / ntg) * (16 * TM);
    const int tn  = (tile % ntg) * (16 * TN);
    const int hf  = lane >> 4;      // lane-half select
    const int l15 = lane & 15;

    // A fragment row m = tm+t*16+l15; halves j -> K = 16*hf + j
    const _Float16* ar[TM];
#pragma unroll
    for (int t = 0; t < TM; ++t)
        ar[t] = A + (size_t)(tm + t * 16 + l15) * lda + hf * 16;
    // B fragment col n = tn+s*16+l15; halves j -> K = 16*hf + j  (W row n holds B column n)
    const _Float16* br[TN];
#pragma unroll
    for (int s = 0; s < TN; ++s)
        br[s] = W + (size_t)(tn + s * 16 + l15) * ldb + hf * 16;

    v8f acc[TM][TN] = {};
    for (int kb = 0; kb < K; kb += 32) {
        v16h af[TM];
        v16h bf[TN];
#pragma unroll
        for (int t = 0; t < TM; ++t) {
            v8h a0 = *(const v8h*)(ar[t] + kb);
            v8h a1 = *(const v8h*)(ar[t] + kb + 8);
#pragma unroll
            for (int j = 0; j < 8; ++j) { af[t][j] = a0[j]; af[t][j + 8] = a1[j]; }
        }
#pragma unroll
        for (int s = 0; s < TN; ++s)
            bf[s] = *(const v16h*)(br[s] + kb);
#pragma unroll
        for (int s = 0; s < TN; ++s)
#pragma unroll
            for (int t = 0; t < TM; ++t)
                acc[t][s] = __builtin_amdgcn_wmma_f32_16x16x32_f16(
                    false, af[t], false, bf[s], (short)0, acc[t][s], false, false);
    }
#pragma unroll
    for (int t = 0; t < TM; ++t) {
#pragma unroll
        for (int s = 0; s < TN; ++s) {
#pragma unroll
            for (int r = 0; r < 8; ++r) {
                int m = tm + t * 16 + r + hf * 8;
                int n = tn + s * 16 + l15;
                size_t o = (size_t)m * N + n;
                float v = acc[t][s][r];
                if constexpr (EP == 0)      { outH[o] = (_Float16)(v + bias[n]); }
                else if constexpr (EP == 1) { float u = v + bias[n]; outH[o] = (_Float16)fmaxf(u, 0.f); }
                else if constexpr (EP == 2) { outF[o] += v + bias[n]; }
                else if constexpr (EP == 3) { float u = v + (bias ? bias[n] : 0.f) + (bias2 ? bias2[n] : 0.f);
                                              outF[o] = u; }
                else if constexpr (EP == 4) { float u = v + bias[n]; outF[o] = u; outH[o] = (_Float16)u; }
                else if constexpr (EP == 5) { outF[o] = bias[n] - 2.f * v; }
            }
        }
    }
}

// ---------------------------------------------------------------------------
// LayerNorm over D=256, one wave per row; optional post-relu; f32 -> f16
// ---------------------------------------------------------------------------
__global__ __launch_bounds__(256)
void ln_kernel(const float* __restrict__ x, const float* __restrict__ g,
               const float* __restrict__ b, _Float16* __restrict__ out, int relu)
{
    const int lane = threadIdx.x & 31;
    const int row  = blockIdx.x * 8 + (threadIdx.x >> 5);
    const float* xr = x + (size_t)row * DM;
    float vals[8], s = 0.f, sq = 0.f;
#pragma unroll
    for (int i = 0; i < 8; ++i) { float v = xr[lane + i * 32]; vals[i] = v; s += v; sq += v * v; }
#pragma unroll
    for (int off = 16; off; off >>= 1) { s += __shfl_xor(s, off); sq += __shfl_xor(sq, off); }
    float mean = s * (1.f / DM);
    float var  = sq * (1.f / DM) - mean * mean;
    float rstd = rsqrtf(var + 1e-5f);
    _Float16* orow = out + (size_t)row * DM;
#pragma unroll
    for (int i = 0; i < 8; ++i) {
        int n = lane + i * 32;
        float v = (vals[i] - mean) * rstd * g[n] + b[n];
        if (relu) v = fmaxf(v, 0.f);
        orow[n] = (_Float16)v;
    }
}

// ---------------------------------------------------------------------------
// small helpers
// ---------------------------------------------------------------------------
__global__ void cvt_copy(const float* __restrict__ src, _Float16* __restrict__ dst,
                         int N, int K, int ldD)
{
    int i = blockIdx.x * blockDim.x + threadIdx.x;
    if (i >= N * K) return;
    int n = i / K, k = i % K;
    dst[(size_t)n * ldD + k] = (_Float16)src[i];
}

__global__ void cvt_pad(const float* __restrict__ src, _Float16* __restrict__ dst,
                        int N, int Ks, int Kd)
{
    int i = blockIdx.x * blockDim.x + threadIdx.x;
    if (i >= N * Kd) return;
    int n = i / Kd, k = i % Kd;
    dst[i] = (k < Ks) ? (_Float16)src[(size_t)n * Ks + k] : (_Float16)0.f;
}

__global__ void cvt_f16(const float* __restrict__ src, _Float16* __restrict__ dst, int n)
{
    int i = blockIdx.x * blockDim.x + threadIdx.x;
    if (i < n) dst[i] = (_Float16)src[i];
}

__global__ void zero_f16(_Float16* dst, int n)
{
    int i = blockIdx.x * blockDim.x + threadIdx.x;
    if (i < n) dst[i] = (_Float16)0.f;
}

__global__ void esq_kernel(const float* __restrict__ cb, float* __restrict__ esq)
{
    int i = blockIdx.x * blockDim.x + threadIdx.x;
    if (i >= NQS * NQV) return;
    const float* e = cb + (size_t)i * DM;
    float s = 0.f;
    for (int d = 0; d < DM; ++d) s += e[d] * e[d];
    esq[i] = s;
}

__global__ void add_pos(const float* __restrict__ zq, const float* __restrict__ pos,
                        float* __restrict__ x)
{
    int i = blockIdx.x * blockDim.x + threadIdx.x;
    if (i < NB * DM) x[i] = zq[i] + pos[i & (DM - 1)];
}

// per-row argmin over 1024 pre-biased distances, one wave per row
__global__ __launch_bounds__(256)
void argmin_kernel(const float* __restrict__ sc, int* __restrict__ idx)
{
    const int lane = threadIdx.x & 31;
    const int row  = blockIdx.x * 8 + (threadIdx.x >> 5);
    const float* r = sc + (size_t)row * NQV;
    float mv = 3.4e38f; int mi = 0;
#pragma unroll 4
    for (int j = 0; j < 32; ++j) {
        int n = lane + j * 32;
        float v = r[n];
        if (v < mv) { mv = v; mi = n; }
    }
#pragma unroll
    for (int off = 16; off; off >>= 1) {
        float v2 = __shfl_xor(mv, off);
        int   i2 = __shfl_xor(mi, off);
        if (v2 < mv || (v2 == mv && i2 < mi)) { mv = v2; mi = i2; }
    }
    if (lane == 0) idx[row] = mi;
}

// RVQ stage update: q = E[idx]; zq += q; r -= q; loss += |r_new|^2; emit code
__global__ __launch_bounds__(256)
void rvq_update(const float* __restrict__ cb, const int* __restrict__ idx,
                float* __restrict__ r, _Float16* __restrict__ r16,
                float* __restrict__ zq, float* __restrict__ lossacc,
                float* __restrict__ codes, int stage)
{
    const int lane = threadIdx.x & 31;
    const int row  = blockIdx.x * 8 + (threadIdx.x >> 5);
    const int id   = idx[row];
    const float* e = cb + (size_t)id * DM;
    float s = 0.f;
#pragma unroll
    for (int i = 0; i < 8; ++i) {
        int n = lane + i * 32;
        size_t o = (size_t)row * DM + n;
        float q  = e[n];
        float rn = r[o] - q;
        r[o]  = rn;
        r16[o] = (_Float16)rn;
        zq[o] += q;
        s += rn * rn;
    }
#pragma unroll
    for (int off = 16; off; off >>= 1) s += __shfl_xor(s, off);
    if (lane == 0) {
        atomicAdd(lossacc, s);
        codes[(size_t)row * NQS + stage] = (float)id;
    }
}

__global__ void finalize_loss(const float* acc, float* out)
{
    out[0] = 1.25f * acc[0] / ((float)NB * (float)DM);
}

// kinematic heads: softplus / tanh / scale, normalize, scatter outputs
__global__ void heads_kernel(const float* __restrict__ hs,
                             const float* __restrict__ vb, const float* __restrict__ kb,
                             const float* __restrict__ dyb,
                             const float* __restrict__ nmean, const float* __restrict__ nstd,
                             const float* __restrict__ nscale,
                             float* __restrict__ out_xn, float* __restrict__ out_v,
                             float* __restrict__ out_k)
{
    int i = blockIdx.x * blockDim.x + threadIdx.x;
    if (i >= NB * 25) return;
    int b = i / 25, t = i % 25;
    const float* h = hs + (size_t)b * 80;
    float sv = h[t] + vb[t];
    float v  = (sv > 20.f) ? sv : log1pf(expf(sv));
    float kp = tanhf(h[25 + t] + kb[t]) * 0.5f;
    float dy = (h[50 + t] + dyb[t]) * 0.01f;
    float ph[3];
    ph[0] = v * 0.2f;            // dx = v*DT
    ph[1] = dy;
    ph[2] = v * kp * 0.2f;       // dyaw
#pragma unroll
    for (int c = 0; c < 3; ++c)
        out_xn[(size_t)i * 3 + c] = (ph[c] - nmean[c]) / (nstd[c] + 1e-8f) / nscale[c];
    out_v[i] = v;
    out_k[i] = kp;
}

// ---------------------------------------------------------------------------
#define LAUNCH_GEMM(EPv, TMv, TNv, Aa, LDA, Ww, LDB, B1, B2, OF, OH, Mm, Nn, Kk) \
    do {                                                                         \
        int _tiles = ((Mm) / (16 * (TMv))) * ((Nn) / (16 * (TNv)));              \
        wmma_gemm<EPv, TMv, TNv><<<(_tiles + 7) / 8, 256, 0, stream>>>(          \
            Aa, LDA, Ww, LDB, B1, B2, OF, OH, Mm, Nn, Kk);                       \
    } while (0)

static inline size_t al256(size_t x) { return (x + 255) & ~(size_t)255; }

extern "C" void kernel_launch(void* const* d_in, const int* in_sizes, int n_in,
                              void* d_out, int out_size, void* d_ws, size_t ws_size,
                              hipStream_t stream)
{
    (void)in_sizes; (void)n_in; (void)out_size; (void)ws_size;
    const float* const* in = (const float* const*)d_in;
    const float* x     = in[0];
    const float* Wp    = in[1];
    const float* bp    = in[2];
    const float* pos   = in[3];
    const float* stkW[2][4]; // [enc/dec][Wqkv,Wo,W1,W2]
    const float* stkB[2][4]; // biases
    const float* stkLN[2][4];
    for (int s = 0; s < 2; ++s) {
        int base = (s == 0) ? 4 : 22;
        stkW[s][0] = in[base + 0];  stkB[s][0] = in[base + 1];   // Wqkv/bqkv
        stkW[s][1] = in[base + 2];  stkB[s][1] = in[base + 3];   // Wo/bo
        stkW[s][2] = in[base + 4];  stkB[s][2] = in[base + 5];   // W1/b1
        stkW[s][3] = in[base + 6];  stkB[s][3] = in[base + 7];   // W2/b2
        stkLN[s][0] = in[base + 8];  stkLN[s][1] = in[base + 9];
        stkLN[s][2] = in[base + 10]; stkLN[s][3] = in[base + 11];
    }
    const float* latg  = in[16];
    const float* latbl = in[17];
    const float* latW  = in[18];
    const float* latb  = in[19];
    const float* cb    = in[20];
    const float* dpos  = in[21];
    const float* vW = in[34], *vb = in[35], *kW = in[36], *kb = in[37];
    const float* dyW = in[38], *dyb = in[39];
    const float* nmean = in[40], *nstd = in[41], *nscale = in[42];

    // ---- workspace carve ----
    char* ws = (char*)d_ws;
    size_t off = 0;
    auto alloc = [&](size_t bytes) { size_t o = off; off = al256(off + bytes); return o; };
    size_t oWp  = alloc((size_t)256 * 96 * 2);
    size_t oWv  = alloc((size_t)4 * 256 * 256 * 2);
    size_t oWo  = alloc((size_t)4 * 256 * 256 * 2);
    size_t oW1  = alloc((size_t)4 * 1024 * 256 * 2);
    size_t oW2  = alloc((size_t)4 * 256 * 1024 * 2);
    size_t oLat = alloc((size_t)256 * 256 * 2);
    size_t oCB  = alloc((size_t)NQS * NQV * DM * 2);
    size_t oHd  = alloc((size_t)80 * 256 * 2);
    size_t oEsq = alloc((size_t)NQS * NQV * 4);
    size_t oA0  = alloc((size_t)NB * 96 * 2);
    size_t oX   = alloc((size_t)NB * DM * 4);
    size_t oSC  = alloc((size_t)NB * NQV * 4);   // scores f32; aliased: ff1 f16 [B,1024], head scores [B,80]
    size_t oZQ  = alloc((size_t)NB * DM * 4);
    size_t oR   = alloc((size_t)NB * DM * 4);
    size_t oH   = alloc((size_t)NB * DM * 2);    // LN out f16 / decoder-out f16
    size_t oT   = alloc((size_t)NB * DM * 2);    // attn-v f16 / residual f16
    size_t oIdx = alloc((size_t)NB * 4);
    size_t oLs  = alloc(256);

    _Float16* wWp  = (_Float16*)(ws + oWp);
    _Float16* wWv  = (_Float16*)(ws + oWv);
    _Float16* wWo  = (_Float16*)(ws + oWo);
    _Float16* wW1  = (_Float16*)(ws + oW1);
    _Float16* wW2  = (_Float16*)(ws + oW2);
    _Float16* wLat = (_Float16*)(ws + oLat);
    _Float16* wCB  = (_Float16*)(ws + oCB);
    _Float16* wHd  = (_Float16*)(ws + oHd);
    float*    esq  = (float*)(ws + oEsq);
    _Float16* a0   = (_Float16*)(ws + oA0);
    float*    xres = (float*)(ws + oX);
    float*    sc   = (float*)(ws + oSC);
    _Float16* ff1  = (_Float16*)(ws + oSC);
    float*    zq   = (float*)(ws + oZQ);
    float*    rF   = (float*)(ws + oR);
    _Float16* hln  = (_Float16*)(ws + oH);
    _Float16* t16  = (_Float16*)(ws + oT);
    int*      idx  = (int*)(ws + oIdx);
    float*    loss = (float*)(ws + oLs);

    float* outF     = (float*)d_out;
    float* out_xn   = outF;                 // [B,25,3]
    float* out_loss = outF + 2457600;       // scalar
    float* out_code = outF + 2457601;       // [B,10]
    float* out_v    = outF + 2785281;       // [B,25]
    float* out_k    = outF + 3604481;       // [B,25]

    // ---- weight conversion to f16 (L2-resident) ----
    cvt_pad<<<(256 * 96 + 255) / 256, 256, 0, stream>>>(Wp, wWp, 256, 75, 96);
    for (int g = 0; g < 4; ++g) {
        int s = g >> 1, l = g & 1;
        cvt_copy<<<(65536 + 255) / 256, 256, 0, stream>>>(
            stkW[s][0] + (size_t)l * 768 * 256 + 512 * 256, wWv + (size_t)g * 65536, 256, 256, 256);
        cvt_copy<<<(65536 + 255) / 256, 256, 0, stream>>>(
            stkW[s][1] + (size_t)l * 65536, wWo + (size_t)g * 65536, 256, 256, 256);
        cvt_copy<<<(262144 + 255) / 256, 256, 0, stream>>>(
            stkW[s][2] + (size_t)l * 262144, wW1 + (size_t)g * 262144, 1024, 256, 256);
        cvt_copy<<<(262144 + 255) / 256, 256, 0, stream>>>(
            stkW[s][3] + (size_t)l * 262144, wW2 + (size_t)g * 262144, 256, 1024, 1024);
    }
    cvt_copy<<<(65536 + 255) / 256, 256, 0, stream>>>(latW, wLat, 256, 256, 256);
    cvt_copy<<<(NQS * NQV * DM + 255) / 256, 256, 0, stream>>>(cb, wCB, NQS * NQV, DM, DM);
    zero_f16<<<(80 * 256 + 255) / 256, 256, 0, stream>>>(wHd, 80 * 256);
    cvt_copy<<<(25 * 256 + 255) / 256, 256, 0, stream>>>(vW,  wHd,            25, 256, 256);
    cvt_copy<<<(25 * 256 + 255) / 256, 256, 0, stream>>>(kW,  wHd + 25 * 256, 25, 256, 256);
    cvt_copy<<<(25 * 256 + 255) / 256, 256, 0, stream>>>(dyW, wHd + 50 * 256, 25, 256, 256);
    esq_kernel<<<(NQS * NQV + 255) / 256, 256, 0, stream>>>(cb, esq);
    cvt_pad<<<(NB * 96 + 255) / 256, 256, 0, stream>>>(x, a0, NB, 75, 96);
    hipMemsetAsync(zq, 0, (size_t)NB * DM * 4, stream);
    hipMemsetAsync(loss, 0, 4, stream);

    const int LNG = NB / 8;

    // ---- encoder ----  h = x@Wp^T + bp + pos
    LAUNCH_GEMM(3, 2, 4, a0, 96, wWp, 96, bp, pos, xres, (_Float16*)nullptr, NB, 256, 96);
    for (int l = 0; l < 2; ++l) {
        int g = l;
        ln_kernel<<<LNG, 256, 0, stream>>>(xres, stkLN[0][0] + l * 256, stkLN[0][1] + l * 256, hln, 0);
        LAUNCH_GEMM(0, 2, 4, hln, 256, wWv + (size_t)g * 65536, 256,
                    stkB[0][0] + (size_t)l * 768 + 512, (const float*)nullptr,
                    (float*)nullptr, t16, NB, 256, 256);
        LAUNCH_GEMM(2, 2, 4, t16, 256, wWo + (size_t)g * 65536, 256,
                    stkB[0][1] + l * 256, (const float*)nullptr, xres, (_Float16*)nullptr, NB, 256, 256);
        ln_kernel<<<LNG, 256, 0, stream>>>(xres, stkLN[0][2] + l * 256, stkLN[0][3] + l * 256, hln, 0);
        LAUNCH_GEMM(1, 2, 4, hln, 256, wW1 + (size_t)g * 262144, 256,
                    stkB[0][2] + l * 1024, (const float*)nullptr, (float*)nullptr, ff1, NB, 1024, 256);
        LAUNCH_GEMM(2, 2, 4, ff1, 1024, wW2 + (size_t)g * 262144, 1024,
                    stkB[0][3] + l * 256, (const float*)nullptr, xres, (_Float16*)nullptr, NB, 256, 1024);
    }

    // ---- latent:  z = relu(LN(x)) @ latW^T + latb ;  r0 = z ----
    ln_kernel<<<LNG, 256, 0, stream>>>(xres, latg, latbl, hln, 1);
    LAUNCH_GEMM(4, 2, 4, hln, 256, wLat, 256, latb, (const float*)nullptr, rF, t16, NB, 256, 256);

    // ---- residual VQ: 10 stages ----
    for (int q = 0; q < NQS; ++q) {
        LAUNCH_GEMM(5, 2, 4, t16, 256, wCB + (size_t)q * NQV * DM, 256,
                    esq + (size_t)q * NQV, (const float*)nullptr, sc, (_Float16*)nullptr, NB, NQV, 256);
        argmin_kernel<<<LNG, 256, 0, stream>>>(sc, idx);
        rvq_update<<<LNG, 256, 0, stream>>>(cb + (size_t)q * NQV * DM, idx, rF, t16, zq, loss, out_code, q);
    }
    finalize_loss<<<1, 1, 0, stream>>>(loss, out_loss);

    // ---- decoder ----
    add_pos<<<(NB * DM + 255) / 256, 256, 0, stream>>>(zq, dpos, xres);
    for (int l = 0; l < 2; ++l) {
        int g = 2 + l;
        ln_kernel<<<LNG, 256, 0, stream>>>(xres, stkLN[1][0] + l * 256, stkLN[1][1] + l * 256, hln, 0);
        LAUNCH_GEMM(0, 2, 4, hln, 256, wWv + (size_t)g * 65536, 256,
                    stkB[1][0] + (size_t)l * 768 + 512, (const float*)nullptr,
                    (float*)nullptr, t16, NB, 256, 256);
        LAUNCH_GEMM(2, 2, 4, t16, 256, wWo + (size_t)g * 65536, 256,
                    stkB[1][1] + l * 256, (const float*)nullptr, xres, (_Float16*)nullptr, NB, 256, 256);
        ln_kernel<<<LNG, 256, 0, stream>>>(xres, stkLN[1][2] + l * 256, stkLN[1][3] + l * 256, hln, 0);
        LAUNCH_GEMM(1, 2, 4, hln, 256, wW1 + (size_t)g * 262144, 256,
                    stkB[1][2] + l * 1024, (const float*)nullptr, (float*)nullptr, ff1, NB, 1024, 256);
        LAUNCH_GEMM(2, 2, 4, ff1, 1024, wW2 + (size_t)g * 262144, 1024,
                    stkB[1][3] + l * 256, (const float*)nullptr, xres, (_Float16*)nullptr, NB, 256, 1024);
    }

    // ---- heads:  [B,80] = hd @ [vW;kW;dyW]^T   then kinematics ----
    cvt_f16<<<(NB * DM + 255) / 256, 256, 0, stream>>>(xres, hln, NB * DM);
    LAUNCH_GEMM(3, 2, 1, hln, 256, wHd, 256, (const float*)nullptr, (const float*)nullptr,
                sc, (_Float16*)nullptr, NB, 80, 256);
    heads_kernel<<<(NB * 25 + 255) / 256, 256, 0, stream>>>(
        sc, vb, kb, dyb, nmean, nstd, nscale, out_xn, out_v, out_k);
}
